// AdaptivePositionAwareAttention_42545946034881
// MI455X (gfx1250) — matrix operand order repaired and checked by reading, not compiled, not run on hardware
//
#include <hip/hip_runtime.h>

#define Bb 4
#define Ss 2048
#define Hh 768
#define BSr (Bb*Ss)

typedef __bf16 bf16;
typedef __attribute__((ext_vector_type(16))) __bf16 v16bf;
typedef __attribute__((ext_vector_type(8)))  __bf16 v8bf;
typedef __attribute__((ext_vector_type(8)))  float  v8f;
typedef __attribute__((ext_vector_type(4)))  unsigned u32x4;
typedef __attribute__((ext_vector_type(8)))  unsigned u32x8;

__device__ __forceinline__ v8f wmma_bf16(v16bf a, v16bf b, v8f c) {
  return __builtin_amdgcn_wmma_f32_16x16x32_bf16(false, a, false, b, (short)0, c, false, false);
}
__device__ __forceinline__ v8f zero8() {
  v8f z;
#pragma unroll
  for (int i = 0; i < 8; i++) z[i] = 0.f;
  return z;
}
// A fragment 16x32 bf16: lane = M (mod 16), lane-group selects K-half.
__device__ __forceinline__ v16bf frag_a(const bf16* p, int ld, int lane) {
  int m = lane & 15, g = (lane >> 4) & 1;
  const bf16* q = p + (size_t)m * ld + g * 8;
  v8bf lo = *(const v8bf*)q;
  v8bf hi = *(const v8bf*)(q + 16);
  v16bf f;
#pragma unroll
  for (int i = 0; i < 8; i++) { f[i] = lo[i]; f[i + 8] = hi[i]; }
  return f;
}
// B fragment 32x16 bf16 from row-major [N,K]: lane = N (mod 16), lane-group selects K-half (same K order as A).
__device__ __forceinline__ v16bf frag_b(const bf16* p, int ld, int lane) {
  int n = lane & 15, g = (lane >> 4) & 1;
  const bf16* q = p + (size_t)n * ld + g * 8;
  v8bf lo = *(const v8bf*)q;
  v8bf hi = *(const v8bf*)(q + 16);
  v16bf f;
#pragma unroll
  for (int i = 0; i < 8; i++) { f[i] = lo[i]; f[i + 8] = hi[i]; }
  return f;
}

// ---- Tensor Data Mover: 1-D contiguous copy global->LDS (elems are 2-byte units).
// Issue from ONE wave only (TDM ignores EXEC). Caller waits tensorcnt + barrier.
__device__ __forceinline__ void tdm_copy_to_lds(unsigned lds_off, const void* src,
                                                unsigned elems) {
  unsigned long long ga = (unsigned long long)(uintptr_t)src;
  u32x4 g0;
  g0[0] = 1u;                                                  // count=1, user mode
  g0[1] = lds_off;                                             // lds_addr [63:32]
  g0[2] = (unsigned)(ga & 0xffffffffu);                        // global_addr[95:64]
  g0[3] = (unsigned)((ga >> 32) & 0x01ffffffu) | (2u << 30);   // addr[120:96], type=2
  u32x8 g1;
  g1[0] = (1u << 16);                         // workgroup_mask=0, data_size=1 (2B)
  g1[1] = (elems & 0xffffu) << 16;            // tensor_dim0[15:0]
  g1[2] = (elems >> 16) | (1u << 16);         // tensor_dim0[31:16], tensor_dim1=1
  g1[3] = (elems & 0xffffu) << 16;            // tile_dim0 = elems
  g1[4] = 0u;                                 // tile_dim1=0, tile_dim2=0
  g1[5] = elems;                              // tensor_dim0_stride[31:0]
  g1[6] = 0u;
  g1[7] = 0u;
  asm volatile("tensor_load_to_lds %0, %1" :: "s"(g0), "s"(g1) : "memory");
}

// ---------------- prep: xb = bf16(x + pos_emb) ----------------
__global__ void prep_x(const float* __restrict__ x, const float* __restrict__ pos,
                       bf16* __restrict__ xb) {
  int i = blockIdx.x * 256 + threadIdx.x;
  if (i < BSr * Hh) xb[i] = (bf16)(x[i] + pos[i % (Ss * Hh)]);
}

// ---------------- convert 4 weight matrices to bf16 ----------------
__global__ void conv_w(const float* __restrict__ a, const float* __restrict__ b,
                       const float* __restrict__ c, const float* __restrict__ d,
                       bf16* __restrict__ oa, bf16* __restrict__ ob,
                       bf16* __restrict__ oc, bf16* __restrict__ od) {
  int i = blockIdx.x * 256 + threadIdx.x;
  const int N = Hh * Hh;
  if (i >= 4 * N) return;
  int w = i / N, j = i % N;
  const float* s = (w == 0) ? a : (w == 1) ? b : (w == 2) ? c : d;
  bf16* o = (w == 0) ? oa : (w == 1) ? ob : (w == 2) ? oc : od;
  o[j] = (bf16)s[j];
}

// ---------------- task scale ----------------
__global__ __launch_bounds__(256) void task_scale_k(
    const float* __restrict__ table, const int* __restrict__ task_id,
    const float* __restrict__ Wt1, const float* __restrict__ bt1,
    const float* __restrict__ Wt2, const float* __restrict__ bt2,
    float* __restrict__ ts) {
  __shared__ float te[64];
  __shared__ float h1[Hh];
  __shared__ float rbuf[256];
  int t = threadIdx.x, b = blockIdx.x;
  if (t < 64) te[t] = table[task_id[b] * 64 + t];
  __syncthreads();
  for (int o = t; o < Hh; o += 256) {
    float s = bt1[o];
    for (int k = 0; k < 64; k++) s += Wt1[o * 64 + k] * te[k];
    h1[o] = fmaxf(s, 0.f);
  }
  __syncthreads();
  float part = 0.f;
  for (int o = t; o < Hh; o += 256) {
    float s = bt2[o];
    for (int k = 0; k < Hh; k++) s += Wt2[o * Hh + k] * h1[k];
    part += 1.f / (1.f + __expf(-s));
  }
  rbuf[t] = part;
  __syncthreads();
  for (int s2 = 128; s2 > 0; s2 >>= 1) {
    if (t < s2) rbuf[t] += rbuf[t + s2];
    __syncthreads();
  }
  if (t == 0) ts[b] = rbuf[0] * (1.f / Hh);
}

// ---------------- content gate ci[b,s] ----------------
__global__ __launch_bounds__(256) void content_k(
    const bf16* __restrict__ xb,
    const float* __restrict__ Wc1, const float* __restrict__ bc1,
    const float* __restrict__ Wc2, const float* __restrict__ bc2,
    const float* __restrict__ Wc3, const float* __restrict__ bc3,
    float* __restrict__ ci) {
  __shared__ float xr[Hh];
  __shared__ float h1[384];
  __shared__ float h2[192];
  __shared__ float rbuf[256];
  int t = threadIdx.x, row = blockIdx.x;
  for (int i = t; i < Hh; i += 256) xr[i] = (float)xb[(size_t)row * Hh + i];
  __syncthreads();
  for (int o = t; o < 384; o += 256) {
    float s = bc1[o];
    for (int k = 0; k < Hh; k++) s += Wc1[o * Hh + k] * xr[k];
    h1[o] = fmaxf(s, 0.f);
  }
  __syncthreads();
  if (t < 192) {
    float s = bc2[t];
    for (int k = 0; k < 384; k++) s += Wc2[t * 384 + k] * h1[k];
    h2[t] = fmaxf(s, 0.f);
  }
  __syncthreads();
  float v = (t < 192) ? Wc3[t] * h2[t] : 0.f;
  rbuf[t] = v;
  __syncthreads();
  for (int s2 = 128; s2 > 0; s2 >>= 1) {
    if (t < s2) rbuf[t] += rbuf[t + s2];
    __syncthreads();
  }
  if (t == 0) ci[row] = 1.f / (1.f + __expf(-(rbuf[0] + bc3[0])));
}

// ---------------- QKV projection GEMM (WMMA + TDM staging) ----------------
__global__ __launch_bounds__(256) void qkv_gemm(
    const bf16* __restrict__ xb,
    const bf16* __restrict__ wq, const bf16* __restrict__ wk, const bf16* __restrict__ wv,
    const float* __restrict__ bq, const float* __restrict__ bk, const float* __restrict__ bv,
    bf16* __restrict__ Qb, bf16* __restrict__ Kb, bf16* __restrict__ VTb) {
  __shared__ bf16 As[16 * Hh];
  int mt = blockIdx.x, wsel = blockIdx.y;
  const bf16* W = (wsel == 0) ? wq : (wsel == 1) ? wk : wv;
  const float* bias = (wsel == 0) ? bq : (wsel == 1) ? bk : bv;
  if (threadIdx.x < 32) {
    tdm_copy_to_lds((unsigned)(uintptr_t)&As[0], xb + (size_t)mt * 16 * Hh, 16 * Hh);
    __builtin_amdgcn_s_wait_tensorcnt(0);
  }
  __syncthreads();
  int lane = threadIdx.x & 31, wave = threadIdx.x >> 5;
  v8f acc[6];
#pragma unroll
  for (int j = 0; j < 6; j++) acc[j] = zero8();
  for (int k0 = 0; k0 < Hh; k0 += 32) {
    v16bf a = frag_a(As + k0, Hh, lane);
#pragma unroll
    for (int j = 0; j < 6; j++) {
      const bf16* bp = W + (size_t)(wave * 96 + j * 16) * Hh + k0;
      acc[j] = wmma_bf16(a, frag_b(bp, Hh, lane), acc[j]);
    }
  }
  int g = lane >> 4, nl = lane & 15;
#pragma unroll
  for (int j = 0; j < 6; j++) {
    int n = wave * 96 + j * 16 + nl;
    float bvv = bias[n];
    if (wsel < 2) {
      bf16* out = (wsel == 0) ? Qb : Kb;
      size_t m0 = (size_t)mt * 16 + g * 8;
#pragma unroll
      for (int r = 0; r < 8; r++) out[(m0 + r) * Hh + n] = (bf16)(acc[j][r] + bvv);
    } else {
      int bb = (mt * 16) >> 11;
      int s0 = ((mt * 16) & 2047) + g * 8;
      v8bf tmp;
#pragma unroll
      for (int r = 0; r < 8; r++) tmp[r] = (bf16)(acc[j][r] + bvv);
      *(v8bf*)(VTb + ((size_t)bb * Hh + n) * Ss + s0) = tmp;  // V transposed [b][h][t]
    }
  }
}

// ---------------- fused attention ----------------
__global__ __launch_bounds__(256) void attn_k(
    const bf16* __restrict__ Qb, const bf16* __restrict__ Kb, const bf16* __restrict__ VTb,
    const float* __restrict__ ts, const float* __restrict__ ci,
    const float* __restrict__ Wf1, const float* __restrict__ bf1,
    const float* __restrict__ Wf2, const float* __restrict__ bf2,
    bf16* __restrict__ Ob) {
  extern __shared__ char smem[];
  float* sc = (float*)smem;                       // 131072 B
  bf16* at = (bf16*)(smem + 131072);              // 65536 B
  bf16* qb = (bf16*)(smem + 131072 + 65536);      // 24576 B
  __shared__ float rbuf[256];
  __shared__ float red2[256];
  __shared__ float rowstat[16];
  __shared__ float crow[16];

  int mt = blockIdx.x, bI = blockIdx.y;
  int tid = threadIdx.x, lane = tid & 31, wave = tid >> 5;
  int g = lane >> 4, nl = lane & 15;

  // stage 16 Q rows into LDS via TDM
  if (tid < 32) {
    tdm_copy_to_lds((unsigned)(uintptr_t)qb, Qb + ((size_t)bI * Ss + mt * 16) * Hh, 16 * Hh);
    __builtin_amdgcn_s_wait_tensorcnt(0);
  }
  __syncthreads();

  // phase 1: base = (Q K^T / sqrt(H)) * exp(-|i-j|/2)  -> sc (LDS)
  const bf16* Kbase = Kb + (size_t)bI * Ss * Hh;
  const float inv = rsqrtf((float)Hh);  // ALPHA = 1
  for (int it = 0; it < 16; it++) {
    int t0 = (wave * 16 + it) * 16;
    v8f acc = zero8();
#pragma unroll
    for (int k0 = 0; k0 < Hh; k0 += 32) {
      v16bf a = frag_a(qb + k0, Hh, lane);
      acc = wmma_bf16(a, frag_b(Kbase + (size_t)t0 * Hh + k0, Hh, lane), acc);
    }
    int j = t0 + nl;
#pragma unroll
    for (int r = 0; r < 8; r++) {
      int lr = g * 8 + r;
      float dist = fabsf((float)(mt * 16 + lr - j));
      sc[lr * Ss + j] = acc[r] * inv * __expf(dist * -0.5f);  // BETA = 2
    }
  }
  __syncthreads();

  // phase 2: row sums of base
  {
    int r = tid >> 4, seg = tid & 15;
    const float* p = sc + r * Ss + seg * 128;
    float s = 0.f;
    for (int i = 0; i < 128; i++) s += p[i];
    red2[tid] = s;
    __syncthreads();
    if (tid < 16) {
      float t2 = 0.f;
      for (int i = 0; i < 16; i++) t2 += red2[tid * 16 + i];
      rowstat[tid] = t2;
    }
    __syncthreads();
  }

  // phase 3: fusion gate -> per-row scalar crow
  float tsb = ts[bI];
  for (int r = 0; r < 16; r++) {
    float m = rowstat[r] * (1.f / Ss);
    float cir = ci[(size_t)bI * Ss + mt * 16 + r];
    float f0 = m, f1 = m * tsb, f2 = m * cir;
    float p = 0.f;
    for (int o = tid; o < Hh; o += 256) {
      float pre = Wf1[o * 3 + 0] * f0 + Wf1[o * 3 + 1] * f1 + Wf1[o * 3 + 2] * f2 + bf1[o];
      p += Wf2[o] * fmaxf(pre, 0.f);
    }
    rbuf[tid] = p;
    __syncthreads();
    for (int s2 = 128; s2 > 0; s2 >>= 1) {
      if (tid < s2) rbuf[tid] += rbuf[tid + s2];
      __syncthreads();
    }
    if (tid == 0) {
      float fw = 1.f / (1.f + __expf(-(rbuf[0] + bf2[0])));
      crow[r] = (1.f - fw) + 0.5f * fw * (tsb + cir);
    }
    __syncthreads();
  }

  // phase 4: softmax(base * crow) -> at (bf16, LDS)
  {
    int r = tid >> 4, seg = tid & 15;
    float cr = crow[r];
    float* p = sc + r * Ss + seg * 128;
    bf16* ap = at + r * Ss + seg * 128;
    float mx = -3.4e38f;
    for (int i = 0; i < 128; i++) mx = fmaxf(mx, p[i] * cr);
    red2[tid] = mx;
    __syncthreads();
    float rm = -3.4e38f;
    for (int i = 0; i < 16; i++) rm = fmaxf(rm, red2[(r << 4) + i]);
    __syncthreads();
    float s = 0.f;
    for (int i = 0; i < 128; i++) {
      float e = __expf(p[i] * cr - rm);
      ap[i] = (bf16)e;
      s += e;
    }
    red2[tid] = s;
    __syncthreads();
    float tot = 0.f;
    for (int i = 0; i < 16; i++) tot += red2[(r << 4) + i];
    float invs = 1.f / tot;
    for (int i = 0; i < 128; i++) ap[i] = (bf16)((float)ap[i] * invs);
    __syncthreads();
  }

  // phase 5: PV GEMM from LDS attn, B operand = V^T[b][h][t]
  const bf16* Vbase = VTb + (size_t)bI * Hh * Ss;
  v8f acc[6];
#pragma unroll
  for (int j = 0; j < 6; j++) acc[j] = zero8();
  for (int k0 = 0; k0 < Ss; k0 += 32) {
    v16bf a = frag_a(at + k0, Ss, lane);
#pragma unroll
    for (int j = 0; j < 6; j++) {
      int h0 = (wave * 6 + j) * 16;
      acc[j] = wmma_bf16(a, frag_b(Vbase + (size_t)h0 * Ss + k0, Ss, lane), acc[j]);
    }
  }
#pragma unroll
  for (int j = 0; j < 6; j++) {
    int h = (wave * 6 + j) * 16 + nl;
    size_t row0 = (size_t)bI * Ss + mt * 16 + g * 8;
#pragma unroll
    for (int r = 0; r < 8; r++) Ob[(row0 + r) * Hh + h] = (bf16)acc[j][r];
  }
}

// ---------------- output projection GEMM (WMMA + TDM staging) ----------------
__global__ __launch_bounds__(256) void proj_gemm(
    const bf16* __restrict__ Ob, const bf16* __restrict__ wo,
    const float* __restrict__ bo, float* __restrict__ out) {
  __shared__ bf16 As[16 * Hh];
  int mt = blockIdx.x;
  if (threadIdx.x < 32) {
    tdm_copy_to_lds((unsigned)(uintptr_t)&As[0], Ob + (size_t)mt * 16 * Hh, 16 * Hh);
    __builtin_amdgcn_s_wait_tensorcnt(0);
  }
  __syncthreads();
  int lane = threadIdx.x & 31, wave = threadIdx.x >> 5;
  v8f acc[6];
#pragma unroll
  for (int j = 0; j < 6; j++) acc[j] = zero8();
  for (int k0 = 0; k0 < Hh; k0 += 32) {
    v16bf a = frag_a(As + k0, Hh, lane);
#pragma unroll
    for (int j = 0; j < 6; j++) {
      const bf16* bp = wo + (size_t)(wave * 96 + j * 16) * Hh + k0;
      acc[j] = wmma_bf16(a, frag_b(bp, Hh, lane), acc[j]);
    }
  }
  int g = lane >> 4, nl = lane & 15;
#pragma unroll
  for (int j = 0; j < 6; j++) {
    int n = wave * 96 + j * 16 + nl;
    float bvv = bo[n];
    size_t m0 = (size_t)mt * 16 + g * 8;
#pragma unroll
    for (int r = 0; r < 8; r++) out[(m0 + r) * Hh + n] = acc[j][r] + bvv;
  }
}

extern "C" void kernel_launch(void* const* d_in, const int* in_sizes, int n_in,
                              void* d_out, int out_size, void* d_ws, size_t ws_size,
                              hipStream_t stream) {
  const float* x   = (const float*)d_in[0];
  const int* task_id = (const int*)d_in[1];
  const float* pos = (const float*)d_in[2];
  const float* Wq  = (const float*)d_in[3];  const float* bq = (const float*)d_in[4];
  const float* Wk  = (const float*)d_in[5];  const float* bk = (const float*)d_in[6];
  const float* Wv  = (const float*)d_in[7];  const float* bv = (const float*)d_in[8];
  const float* Wo  = (const float*)d_in[9];  const float* bo = (const float*)d_in[10];
  const float* table = (const float*)d_in[11];
  const float* Wt1 = (const float*)d_in[12]; const float* bt1 = (const float*)d_in[13];
  const float* Wt2 = (const float*)d_in[14]; const float* bt2 = (const float*)d_in[15];
  const float* Wc1 = (const float*)d_in[16]; const float* bc1 = (const float*)d_in[17];
  const float* Wc2 = (const float*)d_in[18]; const float* bc2 = (const float*)d_in[19];
  const float* Wc3 = (const float*)d_in[20]; const float* bc3 = (const float*)d_in[21];
  const float* Wf1 = (const float*)d_in[22]; const float* bf1 = (const float*)d_in[23];
  const float* Wf2 = (const float*)d_in[24]; const float* bf2 = (const float*)d_in[25];

  char* ws = (char*)d_ws;
  size_t off = 0;
  auto take = [&](size_t bytes) -> char* {
    char* p = ws + off;
    off = (off + bytes + 255) & ~(size_t)255;
    return p;
  };
  bf16* xb  = (bf16*)take((size_t)BSr * Hh * 2);
  bf16* Qb  = (bf16*)take((size_t)BSr * Hh * 2);
  bf16* Kb  = (bf16*)take((size_t)BSr * Hh * 2);
  bf16* VTb = (bf16*)take((size_t)BSr * Hh * 2);
  bf16* Ob  = (bf16*)take((size_t)BSr * Hh * 2);
  bf16* wqb = (bf16*)take((size_t)Hh * Hh * 2);
  bf16* wkb = (bf16*)take((size_t)Hh * Hh * 2);
  bf16* wvb = (bf16*)take((size_t)Hh * Hh * 2);
  bf16* wob = (bf16*)take((size_t)Hh * Hh * 2);
  float* ci = (float*)take((size_t)BSr * 4);
  float* ts = (float*)take(256);

  prep_x<<<(BSr * Hh + 255) / 256, 256, 0, stream>>>(x, pos, xb);
  conv_w<<<(4 * Hh * Hh + 255) / 256, 256, 0, stream>>>(Wq, Wk, Wv, Wo, wqb, wkb, wvb, wob);
  task_scale_k<<<Bb, 256, 0, stream>>>(table, task_id, Wt1, bt1, Wt2, bt2, ts);
  content_k<<<BSr, 256, 0, stream>>>(xb, Wc1, bc1, Wc2, bc2, Wc3, bc3, ci);
  qkv_gemm<<<dim3(BSr / 16, 3), 256, 0, stream>>>(xb, wqb, wkb, wvb, bq, bk, bv, Qb, Kb, VTb);
  attn_k<<<dim3(Ss / 16, Bb), 256, 131072 + 65536 + 24576, stream>>>(
      Qb, Kb, VTb, ts, ci, Wf1, bf1, Wf2, bf2, Ob);
  proj_gemm<<<BSr / 16, 256, 0, stream>>>(Ob, wob, bo, (float*)d_out);
}
